// InformationCleansing_90529320665517
// MI455X (gfx1250) — matrix operand order repaired and checked by reading, not compiled, not run on hardware
//
#include <hip/hip_runtime.h>
#include <hip/hip_bf16.h>

// ---------------------------------------------------------------------------
// MI455X (gfx1250) implementation. All GEMMs + attention use
// v_wmma_f32_16x16x32_bf16 (bf16 A/B, f32 accumulate). Problem working set
// (~30 MB) fits in the 192 MB L2, so everything is staged as bf16 in the
// workspace. GEMM tiles are staged with GLOBAL_LOAD_ASYNC_TO_LDS_B128
// (ASYNCcnt-tracked DMA, no VGPR round trip) into double-buffered LDS so the
// next tile's copy overlaps the current tile's WMMA burst.
// ---------------------------------------------------------------------------

typedef __bf16 bf16_t;
typedef bf16_t bf16x4  __attribute__((ext_vector_type(4)));
typedef bf16_t bf16x8  __attribute__((ext_vector_type(8)));
typedef bf16_t bf16x16 __attribute__((ext_vector_type(16)));
typedef float  f32x8   __attribute__((ext_vector_type(8)));

union Frag16 { bf16x16 v; bf16x8 h[2]; };

__device__ __forceinline__ f32x8 wmma_bf16(const Frag16& a, const Frag16& b, f32x8 c) {
  // (neg_a, A, neg_b, B, c_mod, C, reuse_a, reuse_b)
  return __builtin_amdgcn_wmma_f32_16x16x32_bf16(false, a.v, false, b.v, (short)0, c,
                                                 false, false);
}

// --- CDNA5 async global->LDS DMA helpers -----------------------------------
// Flat addresses of LDS are {shared_aperture, lds_offset}; truncating the
// generic pointer to 32 bits yields the wave-relative LDS byte offset the
// async-load VDST operand wants. INST_OFFSET is added to BOTH the LDS and the
// global address (ISA 15.18.3), so one address pair + offset:16 moves 32B.
__device__ __forceinline__ void async_copy32(void* lds_dst, const void* gsrc) {
  unsigned off = (unsigned)(uintptr_t)lds_dst;
  asm volatile("global_load_async_to_lds_b128 %0, %1, off\n\t"
               "global_load_async_to_lds_b128 %0, %1, off offset:16"
               :: "v"(off), "v"(gsrc) : "memory");
}
__device__ __forceinline__ void async_copy8(void* lds_dst, const void* gsrc) {
  unsigned off = (unsigned)(uintptr_t)lds_dst;
  asm volatile("global_load_async_to_lds_b64 %0, %1, off"
               :: "v"(off), "v"(gsrc) : "memory");
}
__device__ __forceinline__ void wait_async0() {
  asm volatile("s_wait_asynccnt 0x0" ::: "memory");
}

// Problem constants (match reference setup_inputs)
#define BB   2
#define LQ   1024
#define CC   256
#define LK   4096
#define NHD  8
#define DH   32

// LDS row padding (elements): 40*2B = 80B stride keeps every 8/16-element
// chunk 16B-aligned and staggers banks.
#define KP 40

// ---------------------------------------------------------------------------
// Generic WMMA GEMM: out[M,N] = epilogue(A[M,K](lda) * Bt[N,K]^T + bias)
// Block tile 128x128, 8 waves, wave tile 32x64 (2x4 of 16x16 frags).
// Double-buffered LDS, async-to-LDS staging: tile k+1 DMA overlaps tile k's
// 8 WMMAs; one barrier per K-step.
// Epilogue: +bias -> optional relu -> *scale -> optional (resid - val)
// ---------------------------------------------------------------------------
__global__ __launch_bounds__(256)
void gemm_wmma_kernel(const bf16_t* __restrict__ A, int lda,
                      const bf16_t* __restrict__ Bt, int ldb,
                      const float* __restrict__ bias,
                      const float* __restrict__ resid,
                      float* __restrict__ outF, bf16_t* __restrict__ outB,
                      int ldo, int K, int relu, float scale)
{
  __shared__ bf16_t sA[2][128][KP];
  __shared__ bf16_t sB[2][128][KP];

  const int tid   = threadIdx.x;
  const int wave  = tid >> 5, lane = tid & 31;
  const int lhalf = lane >> 4, lm = lane & 15;
  const int waveM = wave & 3, waveN = wave >> 2;     // 4 waves along M, 2 along N
  const int row0  = blockIdx.x * 128, col0 = blockIdx.y * 128;

  f32x8 acc[2][4];
#pragma unroll
  for (int mi = 0; mi < 2; ++mi)
#pragma unroll
    for (int ni = 0; ni < 4; ++ni) acc[mi][ni] = (f32x8){0,0,0,0,0,0,0,0};

  const int sr = tid >> 1, sh = tid & 1;             // staging: 128 rows x 2 halves
  const int c0 = lhalf * 16, c1 = lhalf * 16 + 8; // A-frag K chunk offsets

  const int nk = K >> 5;
  // prologue: DMA tile 0 into buffer 0
  async_copy32(&sA[0][sr][sh * 16], &A[(size_t)(row0 + sr) * lda + sh * 16]);
  async_copy32(&sB[0][sr][sh * 16], &Bt[(size_t)(col0 + sr) * ldb + sh * 16]);

  for (int t = 0; t < nk; ++t) {
    const int cur = t & 1;
    wait_async0();          // this thread's DMAs for tile t are in LDS
    __syncthreads();        // everyone's DMAs landed; prev-tile readers done
    if (t + 1 < nk) {       // overlap tile t+1 DMA with tile t WMMAs
      const int kb = (t + 1) << 5, nb = cur ^ 1;
      async_copy32(&sA[nb][sr][sh * 16], &A[(size_t)(row0 + sr) * lda + kb + sh * 16]);
      async_copy32(&sB[nb][sr][sh * 16], &Bt[(size_t)(col0 + sr) * ldb + kb + sh * 16]);
    }

    Frag16 af[2], bfr[4];
#pragma unroll
    for (int mi = 0; mi < 2; ++mi) {
      const int r = waveM * 32 + mi * 16 + lm;
      af[mi].h[0] = *(const bf16x8*)&sA[cur][r][c0];
      af[mi].h[1] = *(const bf16x8*)&sA[cur][r][c1];
    }
#pragma unroll
    for (int ni = 0; ni < 4; ++ni) {
      const int cc = waveN * 64 + ni * 16 + lm;
      bfr[ni].h[0] = *(const bf16x8*)&sB[cur][cc][lhalf * 16 + 0];
      bfr[ni].h[1] = *(const bf16x8*)&sB[cur][cc][lhalf * 16 + 8];
    }
#pragma unroll
    for (int mi = 0; mi < 2; ++mi)
#pragma unroll
      for (int ni = 0; ni < 4; ++ni)
        acc[mi][ni] = wmma_bf16(af[mi], bfr[ni], acc[mi][ni]);
  }

#pragma unroll
  for (int mi = 0; mi < 2; ++mi) {
#pragma unroll
    for (int ni = 0; ni < 4; ++ni) {
      const int col = col0 + waveN * 64 + ni * 16 + lm;
      const float bs = bias ? bias[col] : 0.0f;
#pragma unroll
      for (int e = 0; e < 8; ++e) {
        const int row = row0 + waveM * 32 + mi * 16 + e + 8 * lhalf;
        float v = acc[mi][ni][e] + bs;
        if (relu) v = fmaxf(v, 0.0f);
        v *= scale;
        const size_t o = (size_t)row * ldo + col;
        if (resid) v = resid[o] - v;
        if (outF) outF[o] = v;
        if (outB) outB[o] = (bf16_t)v;
      }
    }
  }
}

// ---------------------------------------------------------------------------
// Flash attention, D=32: one (b, head, 128-q-row) tile per block, 8 waves,
// each wave owns 16 q rows. K-chunk = 32. q·k^T is a single WMMA per 16x16
// logit tile (K-dim == D == 32). Online softmax via width-16 shfl reductions.
// K tile staged with async-to-LDS DMA; V tile is transposed on the way in so
// it goes through VGPRs. P round-trips through per-wave LDS for the C->A
// fragment relayout (same-wave DS is in-order; explicit wait as insurance).
// ---------------------------------------------------------------------------
__global__ __launch_bounds__(256)
void attn_kernel(const bf16_t* __restrict__ qb, const bf16_t* __restrict__ kbuf,
                 const bf16_t* __restrict__ vbuf, const float* __restrict__ Vflat,
                 bf16_t* __restrict__ outb)
{
  __shared__ bf16_t sk[32][KP];          // K chunk  [kcol][d]
  __shared__ bf16_t svt[32][KP];         // V chunk transposed [d][kcol]
  __shared__ bf16_t sp[8][16][KP];       // per-wave probs scratch
  __shared__ float  sbias[32];

  const int tid   = threadIdx.x;
  const int wave  = tid >> 5, lane = tid & 31;
  const int lhalf = lane >> 4, lm = lane & 15;
  const int qt = blockIdx.x, hh = blockIdx.y, b = blockIdx.z;
  const int hoff  = hh * DH;
  const int qrow0 = qt * 128 + wave * 16;
  const int c0 = lhalf * 16, c1 = lhalf * 16 + 8;

  Frag16 aq;  // q A-fragment (16 rows x K=32); q already scaled by 1/sqrt(D)
  {
    const size_t base = (size_t)(b * LQ + qrow0 + lm) * CC + hoff;
    aq.h[0] = *(const bf16x8*)&qb[base + c0];
    aq.h[1] = *(const bf16x8*)&qb[base + c1];
  }

  f32x8 o0 = (f32x8){0,0,0,0,0,0,0,0};
  f32x8 o1 = (f32x8){0,0,0,0,0,0,0,0};
  float m[8], l[8];
#pragma unroll
  for (int e = 0; e < 8; ++e) { m[e] = -3.0e38f; l[e] = 0.0f; }

  const int skc = tid >> 3, ssg = tid & 7;  // staging: 32 rows x 8 segs of 4

  for (int kb = 0; kb < LK; kb += 32) {
    __syncthreads();
    {
      const size_t gb = (size_t)(b * LK + kb + skc) * CC + hoff + ssg * 4;
      async_copy8(&sk[skc][ssg * 4], &kbuf[gb]);        // K tile: direct DMA
      bf16x4 tv = *(const bf16x4*)&vbuf[gb];            // V tile: transpose in
#pragma unroll
      for (int j = 0; j < 4; ++j) svt[ssg * 4 + j][skc] = tv[j];
      if (tid < 32) sbias[tid] = Vflat[b * LK + kb + tid];
    }
    wait_async0();
    __syncthreads();

    Frag16 bk0, bk1;
    bk0.h[0] = *(const bf16x8*)&sk[lm][lhalf * 16];
    bk0.h[1] = *(const bf16x8*)&sk[lm][lhalf * 16 + 8];
    bk1.h[0] = *(const bf16x8*)&sk[16 + lm][lhalf * 16];
    bk1.h[1] = *(const bf16x8*)&sk[16 + lm][lhalf * 16 + 8];

    const f32x8 z = (f32x8){0,0,0,0,0,0,0,0};
    f32x8 s0 = wmma_bf16(aq, bk0, z);
    f32x8 s1 = wmma_bf16(aq, bk1, z);
    const float b0 = sbias[lm], b1 = sbias[16 + lm];

    float alpha[8];
#pragma unroll
    for (int e = 0; e < 8; ++e) {
      s0[e] += b0; s1[e] += b1;
      float t = fmaxf(s0[e], s1[e]);
#pragma unroll
      for (int off = 1; off < 16; off <<= 1) t = fmaxf(t, __shfl_xor(t, off, 32));
      const float mn = fmaxf(m[e], t);
      alpha[e] = __expf(m[e] - mn);
      m[e] = mn;
      s0[e] = __expf(s0[e] - mn);
      s1[e] = __expf(s1[e] - mn);
      float rs = s0[e] + s1[e];
#pragma unroll
      for (int off = 1; off < 16; off <<= 1) rs += __shfl_xor(rs, off, 32);
      l[e] = l[e] * alpha[e] + rs;
    }
#pragma unroll
    for (int e = 0; e < 8; ++e) { o0[e] *= alpha[e]; o1[e] *= alpha[e]; }

#pragma unroll
    for (int e = 0; e < 8; ++e) {
      const int r = e + 8 * lhalf;
      sp[wave][r][lm]      = (bf16_t)s0[e];
      sp[wave][r][16 + lm] = (bf16_t)s1[e];
    }
    asm volatile("s_wait_dscnt 0x0" ::: "memory");

    Frag16 ap, bv0, bv1;
    ap.h[0]  = *(const bf16x8*)&sp[wave][lm][c0];
    ap.h[1]  = *(const bf16x8*)&sp[wave][lm][c1];
    bv0.h[0] = *(const bf16x8*)&svt[lm][lhalf * 16];
    bv0.h[1] = *(const bf16x8*)&svt[lm][lhalf * 16 + 8];
    bv1.h[0] = *(const bf16x8*)&svt[16 + lm][lhalf * 16];
    bv1.h[1] = *(const bf16x8*)&svt[16 + lm][lhalf * 16 + 8];
    o0 = wmma_bf16(ap, bv0, o0);
    o1 = wmma_bf16(ap, bv1, o1);
  }

#pragma unroll
  for (int e = 0; e < 8; ++e) {
    const int row = qrow0 + e + 8 * lhalf;
    const float inv = 1.0f / l[e];
    const size_t ob = (size_t)(b * LQ + row) * CC + hoff;
    outb[ob + lm]      = (bf16_t)(o0[e] * inv);
    outb[ob + 16 + lm] = (bf16_t)(o1[e] * inv);
  }
}

// ---------------------------------------------------------------------------
// Small prep / tail kernels
// ---------------------------------------------------------------------------
__global__ void mean_kernel(const float* __restrict__ S_SM, float* __restrict__ s) {
  const int idx = blockIdx.x * 256 + threadIdx.x;   // 0..511 = b*256+c
  const int b = idx >> 8, c = idx & 255;
  float acc = 0.0f;
  for (int lq = 0; lq < LQ; ++lq) acc += S_SM[(size_t)(b * LQ + lq) * CC + c];
  s[idx] = acc * (1.0f / (float)LQ);
}

__global__ void fill_xs_kernel(const float* __restrict__ s, bf16_t* __restrict__ x) {
  const int idx = blockIdx.x * 256 + threadIdx.x;   // pixel*256+c
  const int pixel = idx >> 8, c = idx & 255;
  const int b = pixel >> 12;
  x[(size_t)pixel * 512 + c] = (bf16_t)s[b * 256 + c];
}

__global__ void transpose_fq_kernel(const float* __restrict__ f_q, bf16_t* __restrict__ x) {
  __shared__ float tile[32][33];
  const int hw0 = blockIdx.x * 32, cc0 = blockIdx.y * 32, b = blockIdx.z;
  const int tx = threadIdx.x & 31, ty = threadIdx.x >> 5;   // 32 x 8
#pragma unroll
  for (int i = 0; i < 32; i += 8)
    tile[ty + i][tx] = f_q[(size_t)(b * CC + cc0 + ty + i) * LK + hw0 + tx];
  __syncthreads();
#pragma unroll
  for (int i = 0; i < 32; i += 8)
    x[(size_t)(b * LK + hw0 + ty + i) * 512 + 256 + cc0 + tx] = (bf16_t)tile[tx][ty + i];
}

__global__ void concat_qin_kernel(const float* __restrict__ S_QP,
                                  const float* __restrict__ S_SM,
                                  bf16_t* __restrict__ qin) {
  const int idx = blockIdx.x * 256 + threadIdx.x;   // row*512+c
  const int row = idx >> 9, c = idx & 511;
  const float v = (c < 256) ? S_QP[(size_t)row * 256 + c]
                            : S_SM[(size_t)row * 256 + (c - 256)];
  qin[idx] = (bf16_t)v;
}

// W[K,N] f32 -> Wt[N,K] bf16 (transposed so GEMM B staging is contiguous)
__global__ void wcvt_kernel(const float* __restrict__ W, bf16_t* __restrict__ Wt,
                            int K, int N) {
  const int idx = blockIdx.x * 256 + threadIdx.x;
  const int n = idx / K, k = idx - n * K;
  Wt[idx] = (bf16_t)W[(size_t)k * N + n];
}

__device__ __forceinline__ float entropy2(float a, float b) {
  const float mx = fmaxf(a, b);
  const float z0 = __expf(a - mx), z1 = __expf(b - mx);
  const float inv = 1.0f / (z0 + z1);
  const float p0 = fmaxf(z0 * inv, 1e-8f), p1 = fmaxf(z1 * inv, 1e-8f);
  return -(p0 * __logf(p0) + p1 * __logf(p1));
}

// P_hat = h @ qp_W2 + b2 (NC=2) + entropy confidence drop V. One wave/pixel.
__global__ void phat_entropy_kernel(const bf16_t* __restrict__ hbuf,
                                    const float* __restrict__ W2,
                                    const float* __restrict__ b2,
                                    const float* __restrict__ Pin,
                                    float* __restrict__ out_phat,
                                    float* __restrict__ out_V,
                                    float* __restrict__ Vflat) {
  const int wave = threadIdx.x >> 5, lane = threadIdx.x & 31;
  const int pixel = blockIdx.x * 8 + wave;
  const int b = pixel >> 12, hw = pixel & 4095;
  float a0 = 0.0f, a1 = 0.0f;
  for (int c = lane; c < CC; c += 32) {
    const float hv = (float)hbuf[(size_t)pixel * CC + c];
    a0 += hv * W2[c * 2 + 0];
    a1 += hv * W2[c * 2 + 1];
  }
#pragma unroll
  for (int off = 1; off < 32; off <<= 1) {
    a0 += __shfl_xor(a0, off, 32);
    a1 += __shfl_xor(a1, off, 32);
  }
  if (lane == 0) {
    const float p0 = a0 + b2[0], p1 = a1 + b2[1];
    out_phat[(size_t)(b * 2 + 0) * LK + hw] = p0;
    out_phat[(size_t)(b * 2 + 1) * LK + hw] = p1;
    const float v = entropy2(p0, p1)
                  - entropy2(Pin[(size_t)(b * 2 + 0) * LK + hw],
                             Pin[(size_t)(b * 2 + 1) * LK + hw]);
    out_V[(size_t)b * LK + hw] = v;
    Vflat[(size_t)b * LK + hw] = v;
  }
}

// ---------------------------------------------------------------------------
extern "C" void kernel_launch(void* const* d_in, const int* in_sizes, int n_in,
                              void* d_out, int out_size, void* d_ws, size_t ws_size,
                              hipStream_t stream)
{
  const float* S_QP    = (const float*)d_in[0];
  const float* S_SM    = (const float*)d_in[1];
  const float* f_q     = (const float*)d_in[2];
  const float* P       = (const float*)d_in[3];
  const float* qp_W1   = (const float*)d_in[4];
  const float* qp_b1   = (const float*)d_in[5];
  const float* qp_W2   = (const float*)d_in[6];
  const float* qp_b2   = (const float*)d_in[7];
  const float* phiQ_W1 = (const float*)d_in[8];
  const float* phiQ_b1 = (const float*)d_in[9];
  const float* phiQ_W2 = (const float*)d_in[10];
  const float* phiQ_b2 = (const float*)d_in[11];
  const float* Wq = (const float*)d_in[12]; const float* bq = (const float*)d_in[13];
  const float* Wk = (const float*)d_in[14]; const float* bk = (const float*)d_in[15];
  const float* Wv = (const float*)d_in[16]; const float* bv = (const float*)d_in[17];
  const float* Wo = (const float*)d_in[18]; const float* bo = (const float*)d_in[19];
  const float* phi_W = (const float*)d_in[20]; const float* phi_b = (const float*)d_in[21];

  float* out      = (float*)d_out;
  float* out_SIC  = out;                       // [B,Lq,C]  = 524288
  float* out_Phat = out + BB * LQ * CC;        // [B,2,H,W] = 16384
  float* out_V    = out_Phat + BB * 2 * LK;    // [B,1,H,W] = 8192

  // Workspace carve-up (~28.5 MB total)
  char* ws = (char*)d_ws;
  size_t off = 0;
  auto alloc = [&](size_t bytes) -> void* {
    void* p = ws + off;
    off += (bytes + 255) & ~(size_t)255;
    return p;
  };
  float*  s_mean = (float*)alloc(BB * CC * 4);
  bf16_t* xbuf   = (bf16_t*)alloc((size_t)BB * LK * 512 * 2);  // [8192,512] concat(s,f_q)
  bf16_t* hbuf   = (bf16_t*)alloc((size_t)BB * LK * CC * 2);
  bf16_t* qin    = (bf16_t*)alloc((size_t)BB * LQ * 512 * 2);
  bf16_t* t1     = (bf16_t*)alloc((size_t)BB * LQ * CC * 2);
  bf16_t* Qb     = (bf16_t*)alloc((size_t)BB * LQ * CC * 2);
  bf16_t* qproj  = (bf16_t*)alloc((size_t)BB * LQ * CC * 2);
  bf16_t* kproj  = (bf16_t*)alloc((size_t)BB * LK * CC * 2);
  bf16_t* vproj  = (bf16_t*)alloc((size_t)BB * LK * CC * 2);
  bf16_t* attnb  = (bf16_t*)alloc((size_t)BB * LQ * CC * 2);
  bf16_t* nmat   = (bf16_t*)alloc((size_t)BB * LQ * CC * 2);
  float*  Vflat  = (float*)alloc((size_t)BB * LK * 4);
  bf16_t* w_qp1t = (bf16_t*)alloc((size_t)256 * 512 * 2);
  bf16_t* w_pq1t = (bf16_t*)alloc((size_t)256 * 512 * 2);
  bf16_t* w_pq2t = (bf16_t*)alloc((size_t)256 * 256 * 2);
  bf16_t* w_qt   = (bf16_t*)alloc((size_t)256 * 256 * 2);
  bf16_t* w_kt   = (bf16_t*)alloc((size_t)256 * 256 * 2);
  bf16_t* w_vt   = (bf16_t*)alloc((size_t)256 * 256 * 2);
  bf16_t* w_ot   = (bf16_t*)alloc((size_t)256 * 256 * 2);
  bf16_t* w_phit = (bf16_t*)alloc((size_t)256 * 256 * 2);
  (void)in_sizes; (void)n_in; (void)out_size; (void)ws_size;

  // --- prep ---
  mean_kernel<<<2, 256, 0, stream>>>(S_SM, s_mean);
  wcvt_kernel<<<512, 256, 0, stream>>>(qp_W1,   w_qp1t, 512, 256);
  wcvt_kernel<<<512, 256, 0, stream>>>(phiQ_W1, w_pq1t, 512, 256);
  wcvt_kernel<<<256, 256, 0, stream>>>(phiQ_W2, w_pq2t, 256, 256);
  wcvt_kernel<<<256, 256, 0, stream>>>(Wq,    w_qt,   256, 256);
  wcvt_kernel<<<256, 256, 0, stream>>>(Wk,    w_kt,   256, 256);
  wcvt_kernel<<<256, 256, 0, stream>>>(Wv,    w_vt,   256, 256);
  wcvt_kernel<<<256, 256, 0, stream>>>(Wo,    w_ot,   256, 256);
  wcvt_kernel<<<256, 256, 0, stream>>>(phi_W, w_phit, 256, 256);
  fill_xs_kernel<<<8192, 256, 0, stream>>>(s_mean, xbuf);
  transpose_fq_kernel<<<dim3(128, 8, 2), 256, 0, stream>>>(f_q, xbuf);
  concat_qin_kernel<<<4096, 256, 0, stream>>>(S_QP, S_SM, qin);

  // --- QueryPrediction MLP: h = relu(x @ qp_W1 + b1) ---
  gemm_wmma_kernel<<<dim3(64, 2), 256, 0, stream>>>(xbuf, 512, w_qp1t, 512, qp_b1,
      nullptr, nullptr, hbuf, 256, 512, 1, 1.0f);
  phat_entropy_kernel<<<1024, 256, 0, stream>>>(hbuf, qp_W2, qp_b2, P,
      out_Phat, out_V, Vflat);

  // --- phi_Q MLP ---
  gemm_wmma_kernel<<<dim3(16, 2), 256, 0, stream>>>(qin, 512, w_pq1t, 512, phiQ_b1,
      nullptr, nullptr, t1, 256, 512, 1, 1.0f);
  gemm_wmma_kernel<<<dim3(16, 2), 256, 0, stream>>>(t1, 256, w_pq2t, 256, phiQ_b2,
      nullptr, nullptr, Qb, 256, 256, 0, 1.0f);

  // --- projections (1/sqrt(D) folded into q) ---
  gemm_wmma_kernel<<<dim3(16, 2), 256, 0, stream>>>(Qb, 256, w_qt, 256, bq,
      nullptr, nullptr, qproj, 256, 256, 0, 0.17677669529663687f);
  gemm_wmma_kernel<<<dim3(64, 2), 256, 0, stream>>>(xbuf + 256, 512, w_kt, 256, bk,
      nullptr, nullptr, kproj, 256, 256, 0, 1.0f);  // cols 256..511 of x == f_q_flat
  gemm_wmma_kernel<<<dim3(64, 2), 256, 0, stream>>>(xbuf + 256, 512, w_vt, 256, bv,
      nullptr, nullptr, vproj, 256, 256, 0, 1.0f);

  // --- confidence-biased flash attention ---
  attn_kernel<<<dim3(LQ / 128, NHD, BB), 256, 0, stream>>>(qproj, kproj, vproj,
      Vflat, attnb);

  // --- output projection + cleansing ---
  gemm_wmma_kernel<<<dim3(16, 2), 256, 0, stream>>>(attnb, 256, w_ot, 256, bo,
      nullptr, nullptr, nmat, 256, 256, 0, 1.0f);
  gemm_wmma_kernel<<<dim3(16, 2), 256, 0, stream>>>(nmat, 256, w_phit, 256, phi_b,
      S_SM, out_SIC, nullptr, 256, 256, 0, 1.0f);
}
